// MultiHeadAttention_87290915324638
// MI455X (gfx1250) — matrix operand order repaired and checked by reading, not compiled, not run on hardware
//
#include <hip/hip_runtime.h>

// MHA forward for MI455X (gfx1250, wave32).
// All 6 matrix contractions run on v_wmma_f32_16x16x32_bf16 (fp32 accumulate).
// fp32 inputs are converted to bf16 while staging tiles into LDS.
// Staging is load-phase/store-phase split so global loads overlap; fragments
// are preloaded so WMMAs issue back-to-back behind staggered dscnt waits.

typedef __attribute__((ext_vector_type(16))) __bf16 v16bf;
typedef __attribute__((ext_vector_type(8)))  float  v8f;

union Frag { v16bf v; uint4 q[2]; };

__device__ __forceinline__ unsigned short f2bf(float f) {
  unsigned int u = __float_as_uint(f);
  unsigned int r = (u + 0x7FFFu + ((u >> 16) & 1u)) >> 16;  // RNE
  return (unsigned short)r;
}

__device__ __forceinline__ v8f v8f_zero() {
  v8f z; for (int i = 0; i < 8; ++i) z[i] = 0.0f; return z;
}

// Problem dimensions (compile-time).
constexpr int Mdim = 4096;   // B*T
constexpr int Ndim = 1024;   // E
constexpr int Kdim = 1024;   // E

// ---------------------------------------------------------------------------
// GEMM: C[M,N] = A[M,K] @ W[K,N]   (A fp32 or bf16, W fp32; bf16 WMMA compute)
// MODE 1: out_bf16 in [B,H,T,D] split-head layout (for Q/K/V)
// MODE 2: out_f32 row-major + bias (final projection)
// Block: 256 threads (8 waves), tile 256(M) x 64(N), K-step 32.
// Each wave owns 32x64 -> 2 A-frags x 4 B-frags -> 8 WMMAs per K-step.
// ---------------------------------------------------------------------------
template <int ABF16, int MODE>
__global__ void __launch_bounds__(256)
gemm_qkvo(const void* __restrict__ A,
          const float* __restrict__ W,
          unsigned short* __restrict__ out_bf16,
          float* __restrict__ out_f32,
          const float* __restrict__ bias) {
  __shared__ alignas(16) unsigned short a_sh[256 * 32];  // [m][k]
  __shared__ alignas(16) unsigned short b_sh[64 * 32];   // transposed: [n][k]

  const int tid  = threadIdx.x;
  const int wave = tid >> 5;
  const int lane = tid & 31;
  const int lhi  = lane >> 4;   // 0: lanes 0-15, 1: lanes 16-31
  const int lm   = lane & 15;
  const int m0   = blockIdx.y * 256;
  const int n0   = blockIdx.x * 64;

  v8f acc[2][4];
  for (int a = 0; a < 2; ++a)
    for (int nt = 0; nt < 4; ++nt) acc[a][nt] = v8f_zero();

  for (int k0 = 0; k0 < Kdim; k0 += 32) {
    __syncthreads();
    // ---- stage A tile 256x32: load phase then store phase ----
    if constexpr (ABF16) {
      const unsigned short* Ab = (const unsigned short*)A;
      uint4 t[4];
#pragma unroll
      for (int i = 0; i < 4; ++i) {
        int c = i * 256 + tid, row = c >> 2, seg = c & 3;
        t[i] = *(const uint4*)&Ab[(size_t)(m0 + row) * Kdim + k0 + seg * 8];
      }
#pragma unroll
      for (int i = 0; i < 4; ++i) {
        int c = i * 256 + tid, row = c >> 2, seg = c & 3;
        *(uint4*)&a_sh[row * 32 + seg * 8] = t[i];
      }
    } else {
      const float* Af = (const float*)A;
      float4 t[8];
#pragma unroll
      for (int i = 0; i < 8; ++i) {
        int c = i * 256 + tid, row = c >> 3, seg = c & 7;
        t[i] = *(const float4*)&Af[(size_t)(m0 + row) * Kdim + k0 + seg * 4];
      }
#pragma unroll
      for (int i = 0; i < 8; ++i) {
        int c = i * 256 + tid, row = c >> 3, seg = c & 7;
        unsigned short* d = &a_sh[row * 32 + seg * 4];
        d[0] = f2bf(t[i].x); d[1] = f2bf(t[i].y);
        d[2] = f2bf(t[i].z); d[3] = f2bf(t[i].w);
      }
    }
    // ---- stage W tile 32x64, transposed into [n][k] ----
    {
      float4 t[2];
#pragma unroll
      for (int i = 0; i < 2; ++i) {
        int c = i * 256 + tid, krow = c >> 4, seg = c & 15;
        t[i] = *(const float4*)&W[(size_t)(k0 + krow) * Ndim + n0 + seg * 4];
      }
#pragma unroll
      for (int i = 0; i < 2; ++i) {
        int c = i * 256 + tid, krow = c >> 4, seg = c & 15, nb = seg * 4;
        b_sh[(nb + 0) * 32 + krow] = f2bf(t[i].x);
        b_sh[(nb + 1) * 32 + krow] = f2bf(t[i].y);
        b_sh[(nb + 2) * 32 + krow] = f2bf(t[i].z);
        b_sh[(nb + 3) * 32 + krow] = f2bf(t[i].w);
      }
    }
    // ---- prefetch next K-step tiles into cache while we compute ----
    if (k0 + 32 < Kdim) {
      if constexpr (ABF16) {
        const unsigned short* Ab = (const unsigned short*)A;
        __builtin_prefetch(&Ab[(size_t)(m0 + tid) * Kdim + k0 + 32], 0, 3);
      } else {
        const float* Af = (const float*)A;
        __builtin_prefetch(&Af[(size_t)(m0 + tid) * Kdim + k0 + 32], 0, 3);
      }
      if (tid < 32) __builtin_prefetch(&W[(size_t)(k0 + 32 + tid) * Ndim + n0], 0, 3);
    }
    __syncthreads();

    // ---- preload all fragments, then back-to-back WMMAs ----
    Frag af[2], bf[4];
#pragma unroll
    for (int a = 0; a < 2; ++a) {
      const int arow = wave * 32 + a * 16 + lm;   // A: M=lm, K=(e<8?e:e+8)+8*lhi
      af[a].q[0] = *(const uint4*)&a_sh[arow * 32 + 8 * lhi];
      af[a].q[1] = *(const uint4*)&a_sh[arow * 32 + 16 + 8 * lhi];
    }
#pragma unroll
    for (int nt = 0; nt < 4; ++nt) {
      const int ncol = nt * 16 + lm;              // B: N=lm, K=(e<8?e:e+8)+8*lhi
      bf[nt].q[0] = *(const uint4*)&b_sh[ncol * 32 + 8 * lhi];
      bf[nt].q[1] = *(const uint4*)&b_sh[ncol * 32 + 16 + 8 * lhi];
    }
#pragma unroll
    for (int a = 0; a < 2; ++a)
#pragma unroll
      for (int nt = 0; nt < 4; ++nt)
        acc[a][nt] = __builtin_amdgcn_wmma_f32_16x16x32_bf16(
            false, af[a].v, false, bf[nt].v, (short)0, acc[a][nt], false, false);
  }

  // ---- epilogue: C layout M = r + 8*lhi, N = lm ----
#pragma unroll
  for (int a = 0; a < 2; ++a)
#pragma unroll
    for (int nt = 0; nt < 4; ++nt)
#pragma unroll
      for (int r = 0; r < 8; ++r) {
        int m = m0 + wave * 32 + a * 16 + r + 8 * lhi;
        int n = n0 + nt * 16 + lm;
        if constexpr (MODE == 1) {
          // split heads: [B,H,T,D]  (T=2048, H=16, D=64)
          int b = m >> 11, t = m & 2047, h = n >> 6, d = n & 63;
          out_bf16[(((size_t)(b * 16 + h)) * 2048 + t) * 64 + d] = f2bf(acc[a][nt][r]);
        } else {
          out_f32[(size_t)m * Ndim + n] = acc[a][nt][r] + bias[n];
        }
      }
}

// ---------------------------------------------------------------------------
// Causal flash attention. Q/K/V bf16 [BH=32][T=2048][D=64].
// Block: 128 threads (4 waves) -> 64 query rows per (b,h). Key blocks of 32.
// O written bf16 row-major [B*T, E] with cols = h*64+d (ready for out proj).
// ---------------------------------------------------------------------------
__global__ void __launch_bounds__(128)
attn_kernel(const unsigned short* __restrict__ Q,
            const unsigned short* __restrict__ Kd,
            const unsigned short* __restrict__ V,
            unsigned short* __restrict__ O) {
  __shared__ alignas(16) unsigned short q_sh[64 * 64];   // [q][d]
  __shared__ alignas(16) unsigned short k_sh[32 * 64];   // [key][d]
  __shared__ alignas(16) unsigned short vt_sh[64 * 32];  // transposed: [d][key]
  __shared__ alignas(16) unsigned short p_sh[4 * 16 * 32];  // per-wave P tile

  const int tid  = threadIdx.x;
  const int wave = tid >> 5;
  const int lane = tid & 31;
  const int lhi  = lane >> 4;
  const int lm   = lane & 15;
  const int bh   = blockIdx.x;          // 0..31
  const int q0   = blockIdx.y * 64;
  const int b    = bh >> 4;
  const int h    = bh & 15;

  const unsigned short* Qb = Q  + (size_t)bh * 2048 * 64;
  const unsigned short* Kb = Kd + (size_t)bh * 2048 * 64;
  const unsigned short* Vb = V  + (size_t)bh * 2048 * 64;

  // stage Q tile once (load phase, then store phase)
  {
    uint4 t[4];
#pragma unroll
    for (int i = 0; i < 4; ++i) {
      int c = i * 128 + tid, row = c >> 3, seg = c & 7;
      t[i] = *(const uint4*)&Qb[(size_t)(q0 + row) * 64 + seg * 8];
    }
#pragma unroll
    for (int i = 0; i < 4; ++i) {
      int c = i * 128 + tid, row = c >> 3, seg = c & 7;
      *(uint4*)&q_sh[row * 64 + seg * 8] = t[i];
    }
  }
  __syncthreads();

  // Q fragments (held in registers for the whole kernel): 2 K-steps of D=64
  Frag qf[2];
  const int qrow = wave * 16 + lm;
#pragma unroll
  for (int kd = 0; kd < 2; ++kd) {
    qf[kd].q[0] = *(const uint4*)&q_sh[qrow * 64 + kd * 32 + 8 * lhi];
    qf[kd].q[1] = *(const uint4*)&q_sh[qrow * 64 + kd * 32 + 16 + 8 * lhi];
  }

  float m_run[8], l_run[8];
  for (int r = 0; r < 8; ++r) { m_run[r] = -3.0e38f; l_run[r] = 0.0f; }
  v8f oacc[4];
  for (int dt = 0; dt < 4; ++dt) oacc[dt] = v8f_zero();

  const int kend = q0 + 64;  // causal: keys < q0+64
  for (int j0 = 0; j0 < kend; j0 += 32) {
    __syncthreads();
    // stage K and V tiles: all loads issued, then all stores
    uint4 tk[2], tv[2];
#pragma unroll
    for (int i = 0; i < 2; ++i) {
      int c = i * 128 + tid, row = c >> 3, seg = c & 7;
      tk[i] = *(const uint4*)&Kb[(size_t)(j0 + row) * 64 + seg * 8];
      tv[i] = *(const uint4*)&Vb[(size_t)(j0 + row) * 64 + seg * 8];
    }
#pragma unroll
    for (int i = 0; i < 2; ++i) {
      int c = i * 128 + tid, row = c >> 3, seg = c & 7;
      *(uint4*)&k_sh[row * 64 + seg * 8] = tk[i];
      const unsigned short* pr = (const unsigned short*)&tv[i];
#pragma unroll
      for (int e = 0; e < 8; ++e) vt_sh[(seg * 8 + e) * 32 + row] = pr[e];
    }
    // prefetch next key block while this one computes
    if (j0 + 32 < kend && tid < 32) {
      __builtin_prefetch(&Kb[(size_t)(j0 + 32 + tid) * 64], 0, 3);
      __builtin_prefetch(&Vb[(size_t)(j0 + 32 + tid) * 64], 0, 3);
    }
    __syncthreads();

    // ---- scores S = (Q @ K^T) * D^-0.5, causal mask ----
    Frag kf[2][2];  // [ct][kd]  B elem (K=d', N=key) = K[key][d'] (contig row)
#pragma unroll
    for (int ct = 0; ct < 2; ++ct)
#pragma unroll
      for (int kd = 0; kd < 2; ++kd) {
        const int keyn = ct * 16 + lm;
        kf[ct][kd].q[0] = *(const uint4*)&k_sh[keyn * 64 + kd * 32 + 8 * lhi];
        kf[ct][kd].q[1] = *(const uint4*)&k_sh[keyn * 64 + kd * 32 + 16 + 8 * lhi];
      }
    v8f s[2];
#pragma unroll
    for (int ct = 0; ct < 2; ++ct) {
      v8f c0 = v8f_zero();
      c0 = __builtin_amdgcn_wmma_f32_16x16x32_bf16(
          false, qf[0].v, false, kf[ct][0].v, (short)0, c0, false, false);
      c0 = __builtin_amdgcn_wmma_f32_16x16x32_bf16(
          false, qf[1].v, false, kf[ct][1].v, (short)0, c0, false, false);
#pragma unroll
      for (int r = 0; r < 8; ++r) {
        int qi = q0 + wave * 16 + r + 8 * lhi;
        int ki = j0 + ct * 16 + lm;
        s[ct][r] = (ki <= qi) ? c0[r] * 0.125f : -3.0e38f;
      }
    }

    // ---- online softmax (row stats across 16 lanes of each half-wave) ----
#pragma unroll
    for (int r = 0; r < 8; ++r) {
      float mx = fmaxf(s[0][r], s[1][r]);
      for (int off = 8; off >= 1; off >>= 1) mx = fmaxf(mx, __shfl_xor(mx, off, 16));
      float mnew = fmaxf(m_run[r], mx);
      float p0 = __expf(s[0][r] - mnew);
      float p1 = __expf(s[1][r] - mnew);
      s[0][r] = p0; s[1][r] = p1;
      float sum = p0 + p1;
      for (int off = 8; off >= 1; off >>= 1) sum += __shfl_xor(sum, off, 16);
      float alpha = __expf(m_run[r] - mnew);
      l_run[r] = l_run[r] * alpha + sum;
      m_run[r] = mnew;
#pragma unroll
      for (int dt = 0; dt < 4; ++dt) oacc[dt][r] *= alpha;
    }

    // ---- P -> LDS (C layout -> A-fragment layout), then O += P @ V ----
#pragma unroll
    for (int ct = 0; ct < 2; ++ct)
#pragma unroll
      for (int r = 0; r < 8; ++r)
        p_sh[wave * 512 + (r + 8 * lhi) * 32 + ct * 16 + lm] = f2bf(s[ct][r]);
    asm volatile("" ::: "memory");  // keep same-wave LDS stores before loads

    Frag pf, vf[4];
    const int pbase = wave * 512 + lm * 32;
    pf.q[0] = *(const uint4*)&p_sh[pbase + 8 * lhi];
    pf.q[1] = *(const uint4*)&p_sh[pbase + 16 + 8 * lhi];
#pragma unroll
    for (int dt = 0; dt < 4; ++dt) {
      const int dcol = dt * 16 + lm;  // B elem (K=key, N=d): table-order keys
      vf[dt].q[0] = *(const uint4*)&vt_sh[dcol * 32 + 8 * lhi];
      vf[dt].q[1] = *(const uint4*)&vt_sh[dcol * 32 + 16 + 8 * lhi];
    }
#pragma unroll
    for (int dt = 0; dt < 4; ++dt)
      oacc[dt] = __builtin_amdgcn_wmma_f32_16x16x32_bf16(
          false, pf.v, false, vf[dt].v, (short)0, oacc[dt], false, false);
  }

  // ---- normalize and write O (row-major [B*T, E], cols = h*64+d) ----
#pragma unroll
  for (int dt = 0; dt < 4; ++dt)
#pragma unroll
    for (int r = 0; r < 8; ++r) {
      int t = q0 + wave * 16 + r + 8 * lhi;
      int col = h * 64 + dt * 16 + lm;
      float val = oacc[dt][r] / l_run[r];
      O[(size_t)(b * 2048 + t) * 1024 + col] = f2bf(val);
    }
}

// ---------------------------------------------------------------------------
extern "C" void kernel_launch(void* const* d_in, const int* in_sizes, int n_in,
                              void* d_out, int out_size, void* d_ws, size_t ws_size,
                              hipStream_t stream) {
  (void)in_sizes; (void)n_in; (void)out_size; (void)ws_size;
  const float* x  = (const float*)d_in[0];
  const float* Wq = (const float*)d_in[1];
  const float* Wk = (const float*)d_in[2];
  const float* Wv = (const float*)d_in[3];
  const float* Wo = (const float*)d_in[4];
  const float* bo = (const float*)d_in[5];
  float* out = (float*)d_out;

  // workspace: Q,K,V bf16 [BH=32][T=2048][D=64] + O bf16 [4096x1024] = 32 MB
  const size_t QKV = (size_t)32 * 2048 * 64;  // 4,194,304 elems
  unsigned short* Qd = (unsigned short*)d_ws;
  unsigned short* Kd = Qd + QKV;
  unsigned short* Vd = Kd + QKV;
  unsigned short* Od = Vd + QKV;

  dim3 ggrid(16, 16);  // N/64, M/256
  gemm_qkvo<0, 1><<<ggrid, 256, 0, stream>>>(x, Wq, Qd, nullptr, nullptr);
  gemm_qkvo<0, 1><<<ggrid, 256, 0, stream>>>(x, Wk, Kd, nullptr, nullptr);
  gemm_qkvo<0, 1><<<ggrid, 256, 0, stream>>>(x, Wv, Vd, nullptr, nullptr);

  attn_kernel<<<dim3(32, 32), 128, 0, stream>>>(Qd, Kd, Vd, Od);

  gemm_qkvo<1, 2><<<ggrid, 256, 0, stream>>>(Od, Wo, nullptr, out, bo);
}
